// VQModule_13666585936176
// MI455X (gfx1250) — compile-verified, matrix operand order checked by reading, not run on hardware
//
#include <hip/hip_runtime.h>
#include <hip/hip_bf16.h>

// ---------------------------------------------------------------------------
// VQ (vector quantization) for MI455X / gfx1250.
//   x:     (32, 256, 32, 32) f32      -> flat rows N = 32768, C = 256
//   embed: (8192, 256) f32 (unit rows)
// Hot loop: argmin_k ||flat_r - embed_k||^2 = argmin_k (e2[k] - 2*dot(flat_r, embed_k))
// Strategy: fused bf16-split WMMA GEMM + running argmin (never materialize d2).
//  - dual accumulators break the WMMA dependency chain
//  - codebook stored hi/lo-interleaved (1KB/row) -> single base reg, imm offsets
//  - explicit kc-level double buffer of B fragments for load/WMMA overlap
//  - global_prefetch_b8 pulls the next chunk's rows a full chunk ahead
// ---------------------------------------------------------------------------

typedef __bf16 bf16_t;
typedef __attribute__((ext_vector_type(16))) __bf16 v16bf;
typedef __attribute__((ext_vector_type(8)))  __bf16 v8bf;
typedef __attribute__((ext_vector_type(8)))  float  v8f;

#define EMB     256
#define NROWS   32768            // 32 * 32 * 32
#define KD      8192
#define EPSILON 1e-5f
#define XTOT    (NROWS * EMB)    // 8388608
#define EROW    512              // elements per combined codebook row (hi+lo)

// ---- workspace layout (bytes) ----
#define WS_XH   ((size_t)0)
#define WS_XL   (WS_XH + (size_t)NROWS * EMB * 2)
#define WS_EHL  (WS_XL + (size_t)NROWS * EMB * 2)
#define WS_E2   (WS_EHL + (size_t)KD * EROW * 2)
#define WS_IDS  (WS_E2 + (size_t)KD * 4)
#define WS_PART (WS_IDS + (size_t)NROWS * 4)

union Frag16 { v16bf v; v8bf h[2]; };

// ---------------------------------------------------------------------------
// Split x (in NCHW order, coalesced reads) into bf16 hi/lo "flat" layout.
// flat[r][c] with r = (b*32 + w)*32 + h.
// ---------------------------------------------------------------------------
__global__ __launch_bounds__(256) void prep_x_kernel(
    const float* __restrict__ x, bf16_t* __restrict__ xh, bf16_t* __restrict__ xl) {
  int t = blockIdx.x * 256 + threadIdx.x;          // flat index in x order
  int w = t & 31;
  int h = (t >> 5) & 31;
  int c = (t >> 10) & 255;
  int b = t >> 18;
  float v  = x[t];
  int   r  = (b * 32 + w) * 32 + h;
  bf16_t hi = (bf16_t)v;
  float  lo = v - (float)hi;
  xh[r * EMB + c] = hi;
  xl[r * EMB + c] = (bf16_t)lo;
}

// ---------------------------------------------------------------------------
// Split embed into interleaved bf16 hi/lo rows (stride EROW=512 elements:
// [0,256) = hi, [256,512) = lo) and compute e2[k] = ||embed[k]||^2.
// ---------------------------------------------------------------------------
__global__ __launch_bounds__(256) void prep_embed_kernel(
    const float* __restrict__ embed, bf16_t* __restrict__ ehl,
    float* __restrict__ e2) {
  int k = blockIdx.x;
  int c = threadIdx.x;
  float v  = embed[k * EMB + c];
  bf16_t hi = (bf16_t)v;
  ehl[k * EROW + c]       = hi;
  ehl[k * EROW + EMB + c] = (bf16_t)(v - (float)hi);

  __shared__ float red[256];
  red[c] = v * v;
  __syncthreads();
  for (int s = 128; s > 0; s >>= 1) {
    if (c < s) red[c] += red[c + s];
    __syncthreads();
  }
  if (c == 0) e2[k] = red[0];
}

// ---------------------------------------------------------------------------
// Fused GEMM + argmin.
// Grid: 256 workgroups x 256 threads (8 waves). Each wave owns 16 rows.
// A (16x256, hi+lo) stays in registers; loop over 512 chunks of 16 codebook
// entries, 8 K-steps of V_WMMA_F32_16X16X32_BF16 x 3 split terms each.
// ---------------------------------------------------------------------------
__global__ __launch_bounds__(256) void vq_argmin_kernel(
    const bf16_t* __restrict__ xh, const bf16_t* __restrict__ xl,
    const bf16_t* __restrict__ ehl, const float* __restrict__ e2,
    int* __restrict__ ids) {
  const int wave = threadIdx.x >> 5;
  const int lane = threadIdx.x & 31;
  const int r0   = blockIdx.x * 128 + wave * 16;

  // 16-bit A 16x32 layout: lanes 0-15 hold K base 0 (K 0..7 in e0..7, 16..23
  // in e8..15); lanes 16-31 hold K base 8 (K 8..15, 24..31).
  const int arow = r0 + (lane & 15);
  const int koff = (lane < 16) ? 0 : 8;

  v16bf ah[8], al[8];
#pragma unroll
  for (int kc = 0; kc < 8; ++kc) {
    const bf16_t* ph = xh + (size_t)arow * EMB + kc * 32 + koff;
    const bf16_t* pl = xl + (size_t)arow * EMB + kc * 32 + koff;
    Frag16 fh, fl;
    fh.h[0] = *(const v8bf*)(ph);
    fh.h[1] = *(const v8bf*)(ph + 16);
    fl.h[0] = *(const v8bf*)(pl);
    fl.h[1] = *(const v8bf*)(pl + 16);
    ah[kc] = fh.v;
    al[kc] = fl.v;
  }

  float minv[8];
  int   mini[8];
#pragma unroll
  for (int i = 0; i < 8; ++i) { minv[i] = 3.0e38f; mini[i] = 0; }

  const int ncol = lane & 15;
  // Each combined row is 1KB = 8 cachelines; split coverage across the two
  // lane halves (both halves map to the same 16 rows).
  const int pfoff = (lane >> 4) << 9;      // 0 or 512 bytes

  for (int n0 = 0; n0 < KD; n0 += 16) {
    const int brow = n0 + ncol;              // B 32x16 layout mirrors A
    // One base register; every fragment load is an immediate offset off it.
    const bf16_t* pb = ehl + (size_t)brow * EROW + koff;

    // Prefetch next chunk's rows (uniform branch -> EXEC stays all-ones).
    if (n0 + 16 < KD) {
      const char* nb = (const char*)(ehl + (size_t)(brow + 16) * EROW) + pfoff;
      __builtin_prefetch(nb, 0, 0);
      __builtin_prefetch(nb + 128, 0, 0);
      __builtin_prefetch(nb + 256, 0, 0);
      __builtin_prefetch(nb + 384, 0, 0);
    }

    v8f acc0 = {};   // hi*hi chain
    v8f acc1 = {};   // cross-term chain (hi*lo + lo*hi)

    // Explicit double buffer: load kc+1's fragments before kc's WMMAs issue,
    // so each WMMA waits on loads issued a full K-step earlier.
    Frag16 bh[2], bl[2];
    bh[0].h[0] = *(const v8bf*)(pb +   0);
    bh[0].h[1] = *(const v8bf*)(pb +  16);
    bl[0].h[0] = *(const v8bf*)(pb + EMB);
    bl[0].h[1] = *(const v8bf*)(pb + EMB + 16);
#pragma unroll
    for (int kc = 0; kc < 8; ++kc) {
      const int cur = kc & 1;
      const int nxt = cur ^ 1;
      if (kc < 7) {
        const bf16_t* pn = pb + (kc + 1) * 32;
        bh[nxt].h[0] = *(const v8bf*)(pn +   0);
        bh[nxt].h[1] = *(const v8bf*)(pn +  16);
        bl[nxt].h[0] = *(const v8bf*)(pn + EMB);
        bl[nxt].h[1] = *(const v8bf*)(pn + EMB + 16);
      }
      // 3-term split: hi*hi + hi*lo + lo*hi (f32 accumulate) ~ f32 dot.
      acc0 = __builtin_amdgcn_wmma_f32_16x16x32_bf16(
          false, ah[kc], false, bh[cur].v, (short)0, acc0, false, false);
      acc1 = __builtin_amdgcn_wmma_f32_16x16x32_bf16(
          false, ah[kc], false, bl[cur].v, (short)0, acc1, false, false);
      acc1 = __builtin_amdgcn_wmma_f32_16x16x32_bf16(
          false, al[kc], false, bh[cur].v, (short)0, acc1, false, false);
    }

    const float e2n = e2[brow];
#pragma unroll
    for (int vi = 0; vi < 8; ++vi) {
      float s = fmaf(-2.0f, acc0[vi] + acc1[vi], e2n);  // x2 const per row
      if (s < minv[vi]) { minv[vi] = s; mini[vi] = brow; }
    }
  }

  // Per-row argmin reduction: row (vi + 8*(lane/16)) is spread over the 16
  // lanes of a half-wave; butterfly with masks 1,2,4,8 stays inside halves.
#pragma unroll
  for (int vi = 0; vi < 8; ++vi) {
    float mv = minv[vi];
    int   mi = mini[vi];
    for (int off = 1; off < 16; off <<= 1) {
      float ov = __shfl_xor(mv, off, 32);
      int   oi = __shfl_xor(mi, off, 32);
      if (ov < mv || (ov == mv && oi < mi)) { mv = ov; mi = oi; }
    }
    if ((lane & 15) == 0) {
      int row = r0 + vi + ((lane >> 4) << 3);
      ids[row] = mi;
    }
  }
}

// ---------------------------------------------------------------------------
// Epilogue kernels (all bandwidth-trivial vs the GEMM).
// ---------------------------------------------------------------------------
__global__ __launch_bounds__(256) void zero_kernel(float* __restrict__ p, int n) {
  int t = blockIdx.x * 256 + threadIdx.x;
  if (t < n) p[t] = 0.0f;
}

// st_out[b,c,h,w] = x + (embed[id] - x); partial loss per block.
__global__ __launch_bounds__(256) void gather_kernel(
    const float* __restrict__ x, const float* __restrict__ embed,
    const int* __restrict__ ids, float* __restrict__ st_out,
    float* __restrict__ partials) {
  int t = blockIdx.x * 256 + threadIdx.x;      // x order -> coalesced writes
  int w = t & 31;
  int h = (t >> 5) & 31;
  int c = (t >> 10) & 255;
  int b = t >> 18;
  int r  = (b * 32 + w) * 32 + h;
  int id = ids[r];
  float e  = embed[id * EMB + c];
  float xv = x[t];
  st_out[t] = xv + (e - xv);
  float d = e - xv;

  __shared__ float red[256];
  red[threadIdx.x] = d * d;
  __syncthreads();
  for (int s = 128; s > 0; s >>= 1) {
    if (threadIdx.x < s) red[threadIdx.x] += red[threadIdx.x + s];
    __syncthreads();
  }
  if (threadIdx.x == 0) partials[blockIdx.x] = red[0];
}

__global__ __launch_bounds__(256) void loss_final_kernel(
    const float* __restrict__ partials, float* __restrict__ loss) {
  __shared__ float red[256];
  float s = 0.0f;
  for (int i = threadIdx.x; i < 32768; i += 256) s += partials[i];
  red[threadIdx.x] = s;
  __syncthreads();
  for (int st = 128; st > 0; st >>= 1) {
    if (threadIdx.x < st) red[threadIdx.x] += red[threadIdx.x + st];
    __syncthreads();
  }
  if (threadIdx.x == 0) *loss = red[0] / (float)XTOT;
}

// ids_out[b,h,w] = ids[(b*32 + w)*32 + h]
__global__ __launch_bounds__(256) void ids_out_kernel(
    const int* __restrict__ ids, int* __restrict__ out) {
  int t = blockIdx.x * 256 + threadIdx.x;      // 32768
  int w = t & 31;
  int h = (t >> 5) & 31;
  int b = t >> 10;
  out[t] = ids[(b * 32 + w) * 32 + h];
}

__global__ __launch_bounds__(256) void counts_kernel(
    const int* __restrict__ ids, float* __restrict__ tnum) {
  int r = blockIdx.x * 256 + threadIdx.x;      // 32768 rows
  atomicAdd(&tnum[ids[r]], 1.0f);
}

// new_track_enc[k][c] += flat[r][c] for rows with ids[r]==k
__global__ __launch_bounds__(256) void scatter_kernel(
    const float* __restrict__ x, const int* __restrict__ ids,
    float* __restrict__ tenc) {
  int t = blockIdx.x * 256 + threadIdx.x;      // x order
  int w = t & 31;
  int h = (t >> 5) & 31;
  int c = (t >> 10) & 255;
  int b = t >> 18;
  int r  = (b * 32 + w) * 32 + h;
  int id = ids[r];
  atomicAdd(&tenc[id * EMB + c], x[t]);
}

// new_embed = track_enc * factor / (track_num + eps); counts sum == NROWS
// always, so factor is a compile-time constant.
__global__ __launch_bounds__(256) void finalize_kernel(
    const float* __restrict__ tenc, const float* __restrict__ tnum,
    float* __restrict__ new_embed) {
  int t = blockIdx.x * 256 + threadIdx.x;      // KD*EMB
  int k = t >> 8;
  const float factor = 1.0f + EPSILON * (float)KD / (float)NROWS;
  new_embed[t] = tenc[t] * factor / (tnum[k] + EPSILON);
}

// ---------------------------------------------------------------------------
extern "C" void kernel_launch(void* const* d_in, const int* in_sizes, int n_in,
                              void* d_out, int out_size, void* d_ws, size_t ws_size,
                              hipStream_t stream) {
  const float* x     = (const float*)d_in[0];   // 32*256*32*32
  const float* embed = (const float*)d_in[1];   // 8192*256
  // d_in[2] (track_num), d_in[3] (track_enc) are unused by the reference math.

  char* ws = (char*)d_ws;
  bf16_t* xh  = (bf16_t*)(ws + WS_XH);
  bf16_t* xl  = (bf16_t*)(ws + WS_XL);
  bf16_t* ehl = (bf16_t*)(ws + WS_EHL);
  float*  e2  = (float*)(ws + WS_E2);
  int*    ids = (int*)(ws + WS_IDS);
  float*  part = (float*)(ws + WS_PART);

  // d_out regions (flat, in reference return order)
  float* out        = (float*)d_out;
  float* st_out     = out;                                   // 8388608
  float* loss       = out + (size_t)XTOT;                    // 1
  int*   ids_out    = (int*)(out + (size_t)XTOT + 1);        // 32768
  float* new_embed  = out + (size_t)XTOT + 1 + NROWS;        // 2097152
  float* track_num  = new_embed + (size_t)KD * EMB;          // 8192
  float* track_enc  = track_num + KD;                        // 2097152

  // 1) precision-split preprocessing
  prep_x_kernel<<<XTOT / 256, 256, 0, stream>>>(x, xh, xl);
  prep_embed_kernel<<<KD, 256, 0, stream>>>(embed, ehl, e2);

  // 2) the compute-bound core: fused WMMA GEMM + argmin
  vq_argmin_kernel<<<NROWS / 128, 256, 0, stream>>>(xh, xl, ehl, e2, ids);

  // 3) epilogue
  gather_kernel<<<XTOT / 256, 256, 0, stream>>>(x, embed, ids, st_out, part);
  loss_final_kernel<<<1, 256, 0, stream>>>(part, loss);
  ids_out_kernel<<<NROWS / 256, 256, 0, stream>>>(ids, ids_out);

  zero_kernel<<<(KD + 255) / 256, 256, 0, stream>>>(track_num, KD);
  zero_kernel<<<(KD * EMB + 255) / 256, 256, 0, stream>>>(track_enc, KD * EMB);
  counts_kernel<<<NROWS / 256, 256, 0, stream>>>(ids, track_num);
  scatter_kernel<<<XTOT / 256, 256, 0, stream>>>(x, ids, track_enc);
  finalize_kernel<<<(KD * EMB) / 256, 256, 0, stream>>>(track_enc, track_num, new_embed);
}
